// FastSCOFF_31671088840706
// MI455X (gfx1250) — matrix operand order repaired and checked by reading, not compiled, not run on hardware
//
#include <hip/hip_runtime.h>

// ---------------------------------------------------------------------------
// FastSCOFF forward on gfx1250 (MI455X): bf16 WMMA pipeline with hard-argmax
// rule bucketing (selected-GRU-only, 8x FLOP cut) and CDNA5 async global->LDS
// double-buffered GEMM staging (ASYNCcnt path, no VGPR round-trip).
// ---------------------------------------------------------------------------

typedef __bf16 bf16;
typedef __attribute__((ext_vector_type(16))) __bf16 v16bf;
typedef __attribute__((ext_vector_type(8)))  float  v8f;

#define B_   64
#define NH_  64
#define M_   4096      // B*NH
#define D_   512
#define H_   512
#define KD_  64
#define R_   8
#define G3_  1536      // 3*H
#define HEADS_ 4
#define PERM_CAP 4352
#define MAX_TILES 272

// ---- WMMA helper -----------------------------------------------------------
__device__ __forceinline__ v8f wmma_bf(v16bf a, v16bf b, v8f c) {
  return __builtin_amdgcn_wmma_f32_16x16x32_bf16(
      /*neg_a=*/false, a, /*neg_b=*/false, b,
      /*c_mod=*/(short)0, c, /*reuse_a=*/false, /*reuse_b=*/false);
}

// A fragment (16x32, M x K): lane(half=l>>4, m=l&15) holds row m,
// K in [8*half, 8*half+8) and [16+8*half, 16+8*half+8).
__device__ __forceinline__ v16bf ldfragA(const bf16* base, int stride) {
  int lane = threadIdx.x & 31;
  const bf16* p = base + (lane & 15) * stride + ((lane >> 4) << 3);
  union { v16bf v; uint4 u[2]; } r;
  r.u[0] = *(const uint4*)(p);
  r.u[1] = *(const uint4*)(p + 16);
  return r.v;
}

// B fragment (32x16, K x N) loaded from BT (N rows x K): lane(half, n) holds
// column n, K in [8*half, 8*half+8) and [16+8*half, 16+8*half+8).
__device__ __forceinline__ v16bf ldfragB(const bf16* base, int stride) {
  int lane = threadIdx.x & 31;
  const bf16* p = base + (lane & 15) * stride + ((lane >> 4) << 3);
  union { v16bf v; uint4 u[2]; } r;
  r.u[0] = *(const uint4*)(p);
  r.u[1] = *(const uint4*)(p + 16);
  return r.v;
}

// ---- CDNA5 async global->LDS (ISA ch.10 / 08_async_tensor) -----------------
// Generic AS pointer to LDS: low 32 bits are the 0-based LDS byte address
__device__ __forceinline__ unsigned lds_addr32(const void* p) {
  return (unsigned)(unsigned long long)(__UINTPTR_TYPE__)p;
}
__device__ __forceinline__ void async_ld_b128(unsigned lds_dst, const void* gsrc) {
  asm volatile("global_load_async_to_lds_b128 %0, %1, off"
               :: "v"(lds_dst), "v"(gsrc) : "memory");
}
__device__ __forceinline__ void async_ld_b32(unsigned lds_dst, const void* gsrc) {
  asm volatile("global_load_async_to_lds_b32 %0, %1, off"
               :: "v"(lds_dst), "v"(gsrc) : "memory");
}
__device__ __forceinline__ void wait_async0() {
  asm volatile("s_wait_asynccnt 0" ::: "memory");
}

// ---- f32 -> bf16 staging kernels ------------------------------------------
__global__ void k_convert(const float* __restrict__ in, bf16* __restrict__ out,
                          int total, int cols, int ldo) {
  int idx = blockIdx.x * 256 + threadIdx.x;
  if (idx >= total) return;
  int r = idx / cols, c = idx - r * cols;
  out[r * ldo + c] = (bf16)in[idx];
}

// in: K x N row-major f32;  out: N x K row-major bf16 (i.e. W^T)
__global__ void k_transpose(const float* __restrict__ in, bf16* __restrict__ out,
                            int K, int N) {
  int idx = blockIdx.x * 256 + threadIdx.x;
  if (idx >= K * N) return;
  int n = idx / K, k = idx - n * K;
  out[idx] = (bf16)in[k * N + n];
}

// ---- generic WMMA GEMM: C[M,N] = A[M,K] * BT[N,K]^T ------------------------
// Async-DMA double-buffered LDS staging (one s_wait_asynccnt + one barrier per
// 32-wide K step).
#define G_OUT_F32    0
#define G_OUT_BF     1
#define G_OUT_TRBF   2   // write (b, col, m&63) transposed bf16, b=m>>6
#define G_OUT_F32RES 3   // f32 = acc + res

__global__ __launch_bounds__(256) void k_gemm(
    const bf16* __restrict__ A, int lda,
    const bf16* __restrict__ BT, int ldb, int K,
    float* __restrict__ Cf, bf16* __restrict__ Cb, int ldc,
    const float* __restrict__ res, int mode) {
  __shared__ bf16 As[2][64 * 32];
  __shared__ bf16 Bs[2][64 * 32];
  int m0 = blockIdx.y * 64, n0 = blockIdx.x * 64;
  int w = threadIdx.x >> 5, lane = threadIdx.x & 31;
  int mi = w & 3, ni = w >> 2;
  int sr = threadIdx.x >> 2;           // staged row 0..63
  int sc = (threadIdx.x & 3) << 3;     // staged col {0,8,16,24}
  const bf16* gA = A + (size_t)(m0 + sr) * lda + sc;
  const bf16* gB = BT + (size_t)(n0 + sr) * ldb + sc;
  unsigned dA[2] = { lds_addr32(&As[0][sr * 32 + sc]), lds_addr32(&As[1][sr * 32 + sc]) };
  unsigned dB[2] = { lds_addr32(&Bs[0][sr * 32 + sc]), lds_addr32(&Bs[1][sr * 32 + sc]) };
  // prologue: DMA first K tile into buffer 0
  async_ld_b128(dA[0], gA);
  async_ld_b128(dB[0], gB);
  int nk = K >> 5;
  v8f acc0 = {}; v8f acc1 = {};
  for (int i = 0; i < nk; ++i) {
    int cur = i & 1;
    wait_async0();          // this wave's DMA into buf[cur] complete
    __syncthreads();        // all waves' DMA complete; prev compute done
    if (i + 1 < nk) {       // DMA next tile into the other buffer
      async_ld_b128(dA[cur ^ 1], gA + (i + 1) * 32);
      async_ld_b128(dB[cur ^ 1], gB + (i + 1) * 32);
    }
    v16bf a  = ldfragA(&As[cur][(mi * 16) * 32], 32);
    v16bf b0 = ldfragB(&Bs[cur][(ni * 32) * 32], 32);
    v16bf b1 = ldfragB(&Bs[cur][(ni * 32 + 16) * 32], 32);
    acc0 = wmma_bf(a, b0, acc0);
    acc1 = wmma_bf(a, b1, acc1);
  }
  int half = lane >> 4, nn = lane & 15;
#pragma unroll
  for (int i = 0; i < 8; ++i) {
    int m  = m0 + mi * 16 + i + half * 8;
    int na = n0 + ni * 32 + nn;
    int nb = na + 16;
    float v0 = acc0[i], v1 = acc1[i];
    if (mode == G_OUT_F32) {
      Cf[m * ldc + na] = v0; Cf[m * ldc + nb] = v1;
    } else if (mode == G_OUT_BF) {
      Cb[m * ldc + na] = (bf16)v0; Cb[m * ldc + nb] = (bf16)v1;
    } else if (mode == G_OUT_TRBF) {
      int bb = m >> 6, j = m & 63;
      Cb[bb * 32768 + na * 64 + j] = (bf16)v0;
      Cb[bb * 32768 + nb * 64 + j] = (bf16)v1;
    } else {
      Cf[m * ldc + na] = v0 + res[m * ldc + na];
      Cf[m * ldc + nb] = v1 + res[m * ldc + nb];
    }
  }
}

// ---- input attention (per batch): scores -> softmax(+null) -> inp ----------
__global__ __launch_bounds__(256) void k_attn_in(
    const bf16* __restrict__ qb,   // (M,64)
    const bf16* __restrict__ kb,   // (M,64)  (null slot excluded: k_null = 0)
    const bf16* __restrict__ vT,   // (B,512,64)
    bf16* __restrict__ catb) {     // (M,1024), inp written at col 512+
  __shared__ float S[64][65];
  __shared__ bf16  P[64 * 64];
  int b = blockIdx.x;
  int w = threadIdx.x >> 5, lane = threadIdx.x & 31;
  int half = lane >> 4, nn = lane & 15;
  // scores = q @ k^T (64x64), scaled by 1/sqrt(64)
  for (int t = w; t < 16; t += 8) {
    int mi = t >> 2, ni = t & 3;
    v8f acc = {};
#pragma unroll
    for (int k0 = 0; k0 < 64; k0 += 32) {
      v16bf a  = ldfragA(qb + (b * 64 + mi * 16) * 64 + k0, 64);
      v16bf bb = ldfragB(kb + (b * 64 + ni * 16) * 64 + k0, 64);
      acc = wmma_bf(a, bb, acc);
    }
#pragma unroll
    for (int i = 0; i < 8; ++i)
      S[mi * 16 + i + half * 8][ni * 16 + nn] = acc[i] * 0.125f;
  }
  __syncthreads();
  // softmax over 65 slots; null slot has logit 0 and zero value
  if (threadIdx.x < 64) {
    int r = threadIdx.x;
    float mx = 0.f;  // null logit
    for (int n = 0; n < 64; ++n) mx = fmaxf(mx, S[r][n]);
    float den = __expf(-mx);  // null contribution
    for (int n = 0; n < 64; ++n) den += __expf(S[r][n] - mx);
    float inv = 1.f / den;
    for (int n = 0; n < 64; ++n)
      P[r * 64 + n] = (bf16)(__expf(S[r][n] - mx) * inv);
  }
  __syncthreads();
  // inp = P @ v  (64 x 512)
  int mi2 = w & 3, nh = w >> 2;
  for (int nt = 0; nt < 16; ++nt) {
    int n0 = nh * 256 + nt * 16;
    v8f acc = {};
#pragma unroll
    for (int k0 = 0; k0 < 64; k0 += 32) {
      v16bf a  = ldfragA(&P[(mi2 * 16) * 64 + k0], 64);
      v16bf bb = ldfragB(vT + b * 32768 + n0 * 64 + k0, 64);
      acc = wmma_bf(a, bb, acc);
    }
#pragma unroll
    for (int i = 0; i < 8; ++i) {
      int m = b * 64 + mi2 * 16 + i + half * 8;
      catb[m * 1024 + 512 + n0 + nn] = (bf16)acc[i];
    }
  }
}

// ---- rule selection: hard argmax over 8 rule scores ------------------------
__global__ __launch_bounds__(256) void k_select(
    const float* __restrict__ sq,        // (M,64)
    const float* __restrict__ rule_emb,  // (8,64)
    const float* __restrict__ selWk,     // (64,64)
    int* __restrict__ rsel) {
  __shared__ float sk[8 * 64];
  int tid = threadIdx.x;
  for (int idx = tid; idx < 512; idx += 256) {
    int r = idx >> 6, kk = idx & 63;
    float s = 0.f;
    for (int e = 0; e < 64; ++e) s += rule_emb[r * 64 + e] * selWk[e * 64 + kk];
    sk[idx] = s;
  }
  __syncthreads();
  int m = blockIdx.x * 256 + tid;
  float qrow[64];
  const float4* p = (const float4*)(sq + m * 64);
#pragma unroll
  for (int i = 0; i < 16; ++i) {
    float4 f = p[i];
    qrow[4 * i] = f.x; qrow[4 * i + 1] = f.y;
    qrow[4 * i + 2] = f.z; qrow[4 * i + 3] = f.w;
  }
  float best = -3.4e38f; int bi = 0;
  for (int r = 0; r < R_; ++r) {
    float s = 0.f;
#pragma unroll
    for (int kk = 0; kk < 64; ++kk) s += qrow[kk] * sk[r * 64 + kk];
    if (s > best) { best = s; bi = r; }  // first max wins (matches argmax)
  }
  rsel[m] = bi;
}

// ---- bucketing of rows by selected rule ------------------------------------
__global__ void k_binit(int* counts, int* cursor, int* perm) {
  int t = blockIdx.x * 256 + threadIdx.x;
  if (t < R_) { counts[t] = 0; cursor[t] = 0; }
  if (t < PERM_CAP) perm[t] = -1;
}
__global__ void k_bcount(const int* __restrict__ rsel, int* counts) {
  int m = blockIdx.x * 256 + threadIdx.x;
  atomicAdd(&counts[rsel[m]], 1);
}
__global__ void k_bscan(const int* __restrict__ counts, int* poff,
                        int* tileRule, int* tileBase, int* numTiles) {
  if (threadIdx.x != 0 || blockIdx.x != 0) return;
  int off = 0, nt = 0;
  for (int r = 0; r < R_; ++r) {
    poff[r] = off;
    int c = counts[r];
    int t = (c + 15) >> 4;
    for (int j = 0; j < t; ++j) { tileRule[nt] = r; tileBase[nt] = off + j * 16; ++nt; }
    off += t << 4;
  }
  *numTiles = nt;
}
__global__ void k_bscatter(const int* __restrict__ rsel, const int* __restrict__ poff,
                           int* cursor, int* perm) {
  int m = blockIdx.x * 256 + threadIdx.x;
  int r = rsel[m];
  int pos = atomicAdd(&cursor[r], 1);
  perm[poff[r] + pos] = m;
}

// ---- bucketed GRU: only the selected rule per unit -------------------------
__global__ __launch_bounds__(256) void k_gru(
    const bf16* __restrict__ catb,   // (M,1024) = [h_bf | inp_bf]
    const bf16* __restrict__ Wih,    // (8,1536,512) bf16
    const bf16* __restrict__ Whh,
    const float* __restrict__ bih,   // (8,1536)
    const float* __restrict__ bhh,
    const float* __restrict__ hs,    // (M,512) f32
    const int* __restrict__ tileRule, const int* __restrict__ tileBase,
    const int* __restrict__ numTiles, const int* __restrict__ perm,
    float* __restrict__ hnew, bf16* __restrict__ hnewb) {
  int t = blockIdx.x;
  if (t >= *numTiles) return;
  int rule = tileRule[t];
  int base = tileBase[t];
  __shared__ int midx[16];
  __shared__ bf16 Ax[16 * 32];
  __shared__ bf16 Ah[16 * 32];
  if (threadIdx.x < 16) midx[threadIdx.x] = perm[base + threadIdx.x];
  __syncthreads();
  int m0 = midx[0];  // first row of a tile is always valid
  int w = threadIdx.x >> 5, lane = threadIdx.x & 31;
  int half = lane >> 4, nn = lane & 15;
  const bf16* WihR = Wih + (size_t)rule * G3_ * D_;
  const bf16* WhhR = Whh + (size_t)rule * G3_ * H_;
  // per-thread A staging slot (2 bf16 = 4B each for Ah and Ax)
  int sidx = threadIdx.x * 2;
  int srow = sidx >> 5, scol = sidx & 31;
  int smm = midx[srow] >= 0 ? midx[srow] : m0;
  unsigned dAh = lds_addr32(&Ah[srow * 32 + scol]);
  unsigned dAx = lds_addr32(&Ax[srow * 32 + scol]);
  const bf16* gAh = catb + (size_t)smm * 1024 + scol;
  const bf16* gAx = gAh + 512;
  for (int cc = 0; cc < 4; ++cc) {
    int n0 = cc * 128 + w * 16;
    v8f axr = {}, axz = {}, axn = {}, ahr = {}, ahz = {}, ahn = {};
    for (int k0 = 0; k0 < 512; k0 += 32) {
      async_ld_b32(dAh, gAh + k0);   // DMA A tiles straight into LDS
      async_ld_b32(dAx, gAx + k0);
      // prefetch next K step of this lane's weight fragment rows
      if (k0 + 32 < 512)
        __builtin_prefetch(WihR + (size_t)(n0 + (lane & 15)) * 512 + k0 + 32 +
                           ((lane >> 4) << 4), 0, 0);
      wait_async0();
      __syncthreads();
      v16bf ax = ldfragA(Ax, 32);
      v16bf ah = ldfragA(Ah, 32);
      v16bf br = ldfragB(WihR + (size_t)(0    + n0) * 512 + k0, 512);
      v16bf bz = ldfragB(WihR + (size_t)(512  + n0) * 512 + k0, 512);
      v16bf bn = ldfragB(WihR + (size_t)(1024 + n0) * 512 + k0, 512);
      axr = wmma_bf(ax, br, axr);
      axz = wmma_bf(ax, bz, axz);
      axn = wmma_bf(ax, bn, axn);
      br = ldfragB(WhhR + (size_t)(0    + n0) * 512 + k0, 512);
      bz = ldfragB(WhhR + (size_t)(512  + n0) * 512 + k0, 512);
      bn = ldfragB(WhhR + (size_t)(1024 + n0) * 512 + k0, 512);
      ahr = wmma_bf(ah, br, ahr);
      ahz = wmma_bf(ah, bz, ahz);
      ahn = wmma_bf(ah, bn, ahn);
      __syncthreads();
    }
#pragma unroll
    for (int i = 0; i < 8; ++i) {
      int row = i + half * 8;
      int m = midx[row];
      if (m < 0) continue;
      int col = n0 + nn;
      float xr = axr[i] + bih[rule * G3_ + col];
      float xz = axz[i] + bih[rule * G3_ + 512 + col];
      float xn = axn[i] + bih[rule * G3_ + 1024 + col];
      float hr = ahr[i] + bhh[rule * G3_ + col];
      float hz = ahz[i] + bhh[rule * G3_ + 512 + col];
      float hn = ahn[i] + bhh[rule * G3_ + 1024 + col];
      float rg = 1.f / (1.f + __expf(-(xr + hr)));
      float zg = 1.f / (1.f + __expf(-(xz + hz)));
      float ng = tanhf(xn + rg * hn);
      float hp = hs[m * 512 + col];
      float hv = (1.f - zg) * ng + zg * hp;
      hnew[m * 512 + col]  = hv;
      hnewb[m * 512 + col] = (bf16)hv;
    }
  }
}

// ---- communication attention (per batch, 4 heads) --------------------------
__global__ __launch_bounds__(256) void k_comm(
    const bf16* __restrict__ qc,   // (M,128)
    const bf16* __restrict__ kc,   // (M,128)
    const bf16* __restrict__ vcT,  // (B,4,128,64)
    bf16* __restrict__ ctxt) {     // (M,512)
  __shared__ float S[64][65];
  __shared__ bf16  P[64 * 64];
  int b = blockIdx.x;
  int w = threadIdx.x >> 5, lane = threadIdx.x & 31;
  int half = lane >> 4, nn = lane & 15;
  for (int h = 0; h < HEADS_; ++h) {
    for (int t = w; t < 16; t += 8) {
      int mi = t >> 2, ni = t & 3;
      v16bf a  = ldfragA(qc + (b * 64 + mi * 16) * 128 + h * 32, 128);
      v16bf bb = ldfragB(kc + (b * 64 + ni * 16) * 128 + h * 32, 128);
      v8f acc = {};
      acc = wmma_bf(a, bb, acc);
#pragma unroll
      for (int i = 0; i < 8; ++i)
        S[mi * 16 + i + half * 8][ni * 16 + nn] = acc[i] * 0.17677669529663689f;
    }
    __syncthreads();
    if (threadIdx.x < 64) {
      int r = threadIdx.x;
      float mx = -3.4e38f;
      for (int n = 0; n < 64; ++n) mx = fmaxf(mx, S[r][n]);
      float den = 0.f;
      for (int n = 0; n < 64; ++n) den += __expf(S[r][n] - mx);
      float inv = 1.f / den;
      for (int n = 0; n < 64; ++n)
        P[r * 64 + n] = (bf16)(__expf(S[r][n] - mx) * inv);
    }
    __syncthreads();
    int mi2 = w & 3, nb = w >> 2;
    for (int nt = 0; nt < 4; ++nt) {
      int n0 = nb * 64 + nt * 16;
      v8f acc = {};
#pragma unroll
      for (int k0 = 0; k0 < 64; k0 += 32) {
        v16bf a  = ldfragA(&P[(mi2 * 16) * 64 + k0], 64);
        v16bf bb = ldfragB(vcT + b * 32768 + (h * 128 + n0) * 64 + k0, 64);
        acc = wmma_bf(a, bb, acc);
      }
#pragma unroll
      for (int i = 0; i < 8; ++i) {
        int m = b * 64 + mi2 * 16 + i + half * 8;
        ctxt[m * 512 + h * 128 + n0 + nn] = (bf16)acc[i];
      }
    }
    __syncthreads();
  }
}

// ---------------------------------------------------------------------------
extern "C" void kernel_launch(void* const* d_in, const int* in_sizes, int n_in,
                              void* d_out, int out_size, void* d_ws, size_t ws_size,
                              hipStream_t stream) {
  const float* x        = (const float*)d_in[0];
  const float* hs       = (const float*)d_in[1];
  const float* Wq_in    = (const float*)d_in[2];
  const float* Wk_in    = (const float*)d_in[3];
  const float* Wv_in    = (const float*)d_in[4];
  const float* sel_Wq   = (const float*)d_in[5];
  const float* sel_Wk   = (const float*)d_in[6];
  const float* rule_emb = (const float*)d_in[7];
  const float* W_ih     = (const float*)d_in[8];
  const float* W_hh     = (const float*)d_in[9];
  const float* b_ih     = (const float*)d_in[10];
  const float* b_hh     = (const float*)d_in[11];
  const float* Wq_c     = (const float*)d_in[12];
  const float* Wk_c     = (const float*)d_in[13];
  const float* Wv_c     = (const float*)d_in[14];
  const float* Wo_c     = (const float*)d_in[15];
  float* out = (float*)d_out;

  // workspace bump allocator (256B aligned)
  char* base = (char*)d_ws;
  size_t off = 0;
  auto alloc = [&](size_t bytes) -> char* {
    char* p = base + off;
    off = (off + bytes + 255) & ~(size_t)255;
    return p;
  };
  bf16* cat_bf  = (bf16*)alloc((size_t)M_ * 1024 * 2); // [hs_bf | inp_bf]
  bf16* x_bf    = (bf16*)alloc((size_t)M_ * 512 * 2);
  bf16* WqT     = (bf16*)alloc(64 * 512 * 2);
  bf16* WkT     = (bf16*)alloc(64 * 512 * 2);
  bf16* WvT     = (bf16*)alloc(512 * 512 * 2);
  bf16* selWqT  = (bf16*)alloc(64 * 1024 * 2);
  bf16* WqcT    = (bf16*)alloc(128 * 512 * 2);
  bf16* WkcT    = (bf16*)alloc(128 * 512 * 2);
  bf16* WvcT    = (bf16*)alloc(512 * 512 * 2);
  bf16* WocT    = (bf16*)alloc(512 * 512 * 2);
  bf16* Wih_bf  = (bf16*)alloc((size_t)R_ * G3_ * 512 * 2);
  bf16* Whh_bf  = (bf16*)alloc((size_t)R_ * G3_ * 512 * 2);
  bf16* q_bf    = (bf16*)alloc((size_t)M_ * 64 * 2);
  bf16* k_bf    = (bf16*)alloc((size_t)M_ * 64 * 2);
  bf16* vT_bf   = (bf16*)alloc((size_t)M_ * 512 * 2);
  float* sq     = (float*)alloc((size_t)M_ * 64 * 4);
  int*  rsel    = (int*)alloc(M_ * 4);
  int*  meta    = (int*)alloc((32 + 2 * MAX_TILES + PERM_CAP) * 4);
  float* hnew   = (float*)alloc((size_t)M_ * 512 * 4);
  bf16* hnew_bf = (bf16*)alloc((size_t)M_ * 512 * 2);
  bf16* qc_bf   = (bf16*)alloc((size_t)M_ * 128 * 2);
  bf16* kc_bf   = (bf16*)alloc((size_t)M_ * 128 * 2);
  bf16* vcT_bf  = (bf16*)alloc((size_t)M_ * 512 * 2);
  bf16* ctxt_bf = (bf16*)alloc((size_t)M_ * 512 * 2);
  int* counts   = meta;
  int* cursor   = meta + 8;
  int* poff     = meta + 16;
  int* numTiles = meta + 24;
  int* tileRule = meta + 32;
  int* tileBase = meta + 32 + MAX_TILES;
  int* perm     = meta + 32 + 2 * MAX_TILES;
  (void)ws_size; (void)in_sizes; (void)n_in; (void)out_size;

  // ---- stage 0: bf16 conversions / weight transposes ----
  k_convert<<<(M_ * 512 + 255) / 256, 256, 0, stream>>>(hs, cat_bf, M_ * 512, 512, 1024);
  k_convert<<<(M_ * 512 + 255) / 256, 256, 0, stream>>>(x, x_bf, M_ * 512, 512, 512);
  k_convert<<<(R_ * G3_ * 512 + 255) / 256, 256, 0, stream>>>(W_ih, Wih_bf, R_ * G3_ * 512, 512, 512);
  k_convert<<<(R_ * G3_ * 512 + 255) / 256, 256, 0, stream>>>(W_hh, Whh_bf, R_ * G3_ * 512, 512, 512);
  k_transpose<<<(512 * 64 + 255) / 256, 256, 0, stream>>>(Wq_in, WqT, 512, 64);
  k_transpose<<<(512 * 64 + 255) / 256, 256, 0, stream>>>(Wk_in, WkT, 512, 64);
  k_transpose<<<(512 * 512 + 255) / 256, 256, 0, stream>>>(Wv_in, WvT, 512, 512);
  k_transpose<<<(1024 * 64 + 255) / 256, 256, 0, stream>>>(sel_Wq, selWqT, 1024, 64);
  k_transpose<<<(512 * 128 + 255) / 256, 256, 0, stream>>>(Wq_c, WqcT, 512, 128);
  k_transpose<<<(512 * 128 + 255) / 256, 256, 0, stream>>>(Wk_c, WkcT, 512, 128);
  k_transpose<<<(512 * 512 + 255) / 256, 256, 0, stream>>>(Wv_c, WvcT, 512, 512);
  k_transpose<<<(512 * 512 + 255) / 256, 256, 0, stream>>>(Wo_c, WocT, 512, 512);

  // ---- stage 1: input attention projections ----
  k_gemm<<<dim3(1, 64), 256, 0, stream>>>(cat_bf, 1024, WqT, 512, 512,
                                          nullptr, q_bf, 64, nullptr, G_OUT_BF);
  k_gemm<<<dim3(1, 64), 256, 0, stream>>>(x_bf, 512, WkT, 512, 512,
                                          nullptr, k_bf, 64, nullptr, G_OUT_BF);
  k_gemm<<<dim3(8, 64), 256, 0, stream>>>(x_bf, 512, WvT, 512, 512,
                                          nullptr, vT_bf, 0, nullptr, G_OUT_TRBF);
  k_attn_in<<<B_, 256, 0, stream>>>(q_bf, k_bf, vT_bf, cat_bf);

  // ---- stage 2: rule selection (hard argmax) ----
  k_gemm<<<dim3(1, 64), 256, 0, stream>>>(cat_bf, 1024, selWqT, 1024, 1024,
                                          sq, nullptr, 64, nullptr, G_OUT_F32);
  k_select<<<M_ / 256, 256, 0, stream>>>(sq, rule_emb, sel_Wk, rsel);
  k_binit<<<(PERM_CAP + 255) / 256, 256, 0, stream>>>(counts, cursor, perm);
  k_bcount<<<M_ / 256, 256, 0, stream>>>(rsel, counts);
  k_bscan<<<1, 32, 0, stream>>>(counts, poff, tileRule, tileBase, numTiles);
  k_bscatter<<<M_ / 256, 256, 0, stream>>>(rsel, poff, cursor, perm);

  // ---- stage 3: bucketed GRU (selected rule only) ----
  k_gru<<<MAX_TILES, 256, 0, stream>>>(cat_bf, Wih_bf, Whh_bf, b_ih, b_hh, hs,
                                       tileRule, tileBase, numTiles, perm,
                                       hnew, hnew_bf);

  // ---- stage 4: communication attention ----
  k_gemm<<<dim3(2, 64), 256, 0, stream>>>(hnew_bf, 512, WqcT, 512, 512,
                                          nullptr, qc_bf, 128, nullptr, G_OUT_BF);
  k_gemm<<<dim3(2, 64), 256, 0, stream>>>(hnew_bf, 512, WkcT, 512, 512,
                                          nullptr, kc_bf, 128, nullptr, G_OUT_BF);
  k_gemm<<<dim3(8, 64), 256, 0, stream>>>(hnew_bf, 512, WvcT, 512, 512,
                                          nullptr, vcT_bf, 0, nullptr, G_OUT_TRBF);
  k_comm<<<B_, 256, 0, stream>>>(qc_bf, kc_bf, vcT_bf, ctxt_bf);

  // ---- stage 5: output projection + residual ----
  k_gemm<<<dim3(8, 64), 256, 0, stream>>>(ctxt_bf, 512, WocT, 512, 512,
                                          out, nullptr, 512, hnew, G_OUT_F32RES);
}
